// PCNN_39651138077251
// MI455X (gfx1250) — matrix operand order repaired and checked by reading, not compiled, not run on hardware
//
#include <hip/hip_runtime.h>
#include <hip/hip_bf16.h>
#include <stdint.h>

// ---------------------------------------------------------------------------
// DGCNN edge-conv stack for MI455X (gfx1250, wave32, bf16 WMMA, f32 accum)
//  - point-major activations -> contiguous b128 loads for WMMA B operand
//  - weights pre-packed per layer into a contiguous-k A-fragment bf16 layout
//  - MT output tiles per wave: B fragment built once, reused by MT WMMAs
//  - compile-time channel counts: no bounds checks, no divergence near WMMA
// ---------------------------------------------------------------------------

typedef __attribute__((ext_vector_type(16))) __bf16 v16bf;
typedef __attribute__((ext_vector_type(8)))  float  v8f;

union BF16x16 { v16bf v; unsigned short s[16]; uint4 q[2]; };

#define NPTS   1024
#define KNN    20
#define NB     16
#define NPIX   (NPTS * KNN)        // 20480 pixels per batch
#define NEGINF (-3.402823466e38f)

__device__ __forceinline__ unsigned short f32_to_bf16(float f) {
    union { float f; uint32_t u; } c; c.f = f;
    uint32_t u = c.u;
    return (unsigned short)((u + 0x7FFFu + ((u >> 16) & 1u)) >> 16);  // RNE
}

// monotonic float<->uint encoding so atomicMax(uint) == float max
__device__ __forceinline__ unsigned enc_f32(float f) {
    union { float f; unsigned u; } c; c.f = f;
    return (c.u & 0x80000000u) ? ~c.u : (c.u | 0x80000000u);
}
__device__ __forceinline__ float dec_f32(unsigned e) {
    union { float f; unsigned u; } c;
    c.u = (e & 0x80000000u) ? (e ^ 0x80000000u) : ~e;
    return c.f;
}

// ---------------------------------------------------------------------------
// layer-0 input transpose: x(B,C=4,N) -> xT(B,N,C=4) point-major
__global__ void transpose_x0(const float* __restrict__ x, float* __restrict__ xT) {
    int t = blockIdx.x * blockDim.x + threadIdx.x;    // (b*1024+n)*4+c
    int c = t & 3, n = (t >> 2) & 1023, b = t >> 12;
    xT[t] = x[((size_t)b * 4 + c) * NPTS + n];
}

// per-point squared norms (point-major, contiguous)
template <int CIN>
__global__ void sqnorm_kernel(const float* __restrict__ xT, float* __restrict__ sqn) {
    int t = blockIdx.x * blockDim.x + threadIdx.x;    // b*1024+n
    const float4* row = (const float4*)(xT + (size_t)t * CIN);
    float s = 0.f;
#pragma unroll
    for (int q = 0; q < CIN / 4; ++q) {
        float4 a = row[q];
        s += a.x * a.x + a.y * a.y + a.z * a.z + a.w * a.w;
    }
    sqn[t] = s;
}

// per-point top-k (largest -d2 == smallest d2); includes self (d2 == 0)
template <int CIN>
__global__ void knn_kernel(const float* __restrict__ xT, const float* __restrict__ sqn,
                           int* __restrict__ idx) {
    int t = blockIdx.x * blockDim.x + threadIdx.x;
    int b = t >> 10, n = t & 1023;
    const float* xb = xT + (size_t)b * NPTS * CIN;
    const float4* xn4 = (const float4*)(xb + (size_t)n * CIN);
    float best[KNN]; int bidx[KNN];
#pragma unroll
    for (int i = 0; i < KNN; ++i) { best[i] = NEGINF; bidx[i] = n; }
    float worst = NEGINF; int wslot = 0;
    float sn = sqn[t];
    for (int m = 0; m < NPTS; ++m) {
        const float4* xm4 = (const float4*)(xb + (size_t)m * CIN);
        float dot = 0.f;
#pragma unroll
        for (int q = 0; q < CIN / 4; ++q) {
            float4 a = xn4[q], c = xm4[q];
            dot += a.x * c.x + a.y * c.y + a.z * c.z + a.w * c.w;
        }
        float negd2 = 2.f * dot - sn - sqn[b * NPTS + m];
        if (negd2 > worst) {
            best[wslot] = negd2; bidx[wslot] = m;
            worst = best[0]; wslot = 0;
#pragma unroll
            for (int i = 1; i < KNN; ++i)
                if (best[i] < worst) { worst = best[i]; wslot = i; }
        }
    }
    for (int i = 0; i < KNN; ++i) idx[t * KNN + i] = bidx[i];
}

// ---------------------------------------------------------------------------
// pre-pack W(Cout x 2C) into the 16-bit A 16x32 WMMA fragment layout (bf16).
// Fragment (mtile, kc): 32 lanes x 8 dwords. lane L: M=L%16, kbase=(L/16)*16;
// element h -> K = kbase+h (contiguous)
template <int CIN, int COUT>
__global__ void pack_w(const float* __restrict__ W, unsigned* __restrict__ apack) {
    constexpr int TWOC = 2 * CIN;
    constexpr int KCH  = (TWOC + 31) / 32;
    int t = blockIdx.x * blockDim.x + threadIdx.x;
    if (t >= (COUT / 16) * KCH * 32) return;
    int lane = t & 31, rem = t >> 5;
    int kc = rem % KCH, mtile = rem / KCH;
    int m = lane & 15, akbase = (lane >> 4) * 8;
    int o = mtile * 16 + m;
    BF16x16 A;
#pragma unroll
    for (int h = 0; h < 16; ++h) {
        int v = h >> 1, w = h & 1;
        int K = 2 * akbase + h;
        int cf = kc * 32 + K;
        float val = (cf < TWOC) ? W[o * TWOC + cf] : 0.0f;
        A.s[h] = f32_to_bf16(val);
    }
    uint4* dst = (uint4*)apack + (size_t)t * 2;
    dst[0] = A.q[0];
    dst[1] = A.q[1];
}

// ---------------------------------------------------------------------------
// mode 0: accumulate per-channel sum/sumsq (BN stats, pre-activation)
// mode 1: BN + LeakyReLU + max-over-k into point-major encoded buffer
// mode 2: BN + LeakyReLU + global max into encoded output
//
// One wave computes MT 16x16 output tiles for the same 16 pixels: the gathered
// B fragment is built once per K-chunk and reused by MT back-to-back WMMAs
// (independent accumulators -> latency hiding, 1/MT the gather traffic).
template <int CIN, int COUT, int MT>
__global__ __launch_bounds__(32)
void edgeconv_kernel(const float* __restrict__ xT, const int* __restrict__ idx,
                     const unsigned* __restrict__ apack,
                     const float* __restrict__ scA, const float* __restrict__ scC,
                     float* __restrict__ sums,
                     unsigned* __restrict__ hEnc, unsigned* __restrict__ outEnc,
                     int mode) {
    constexpr int KCH = (2 * CIN + 31) / 32;
    int lane   = threadIdx.x;
    int ptile  = blockIdx.x;           // 0..1279
    int mtile0 = blockIdx.y * MT;
    int b      = blockIdx.z;

    int pix = ptile * 16 + (lane & 15);
    int n   = pix / KNN;
    int kk  = pix % KNN;
    int j   = idx[(b * NPTS + n) * KNN + kk];

    const float* xb = xT + (size_t)b * NPTS * CIN;
    const float* xn = xb + (size_t)n * CIN;
    const float* xj = xb + (size_t)j * CIN;

    int bkbase = (lane >> 4) * 16;
    const uint4* apLane = (const uint4*)apack + lane * 2;

    v8f acc[MT];
#pragma unroll
    for (int t = 0; t < MT; ++t) acc[t] = (v8f){};

    if constexpr (CIN == 4) {
        // single chunk: cf 0..3 = xj-xn, 4..7 = xn, 8..31 = 0
        BF16x16 Bt;
#pragma unroll
        for (int h = 0; h < 16; ++h) Bt.s[h] = 0;
        if (bkbase == 0) {
            float4 aj = *(const float4*)xj;
            float4 an = *(const float4*)xn;
            Bt.s[0] = f32_to_bf16(aj.x - an.x);
            Bt.s[1] = f32_to_bf16(aj.y - an.y);
            Bt.s[2] = f32_to_bf16(aj.z - an.z);
            Bt.s[3] = f32_to_bf16(aj.w - an.w);
            Bt.s[4] = f32_to_bf16(an.x);
            Bt.s[5] = f32_to_bf16(an.y);
            Bt.s[6] = f32_to_bf16(an.z);
            Bt.s[7] = f32_to_bf16(an.w);
        }
#pragma unroll
        for (int t = 0; t < MT; ++t) {
            const uint4* ap = apLane + (size_t)((mtile0 + t) * KCH) * 64;
            BF16x16 A;  A.q[0] = ap[0];  A.q[1] = ap[1];
            acc[t] = __builtin_amdgcn_wmma_f32_16x16x32_bf16(
                         false, A.v, false, Bt.v, (short)0, acc[t], false, false);
        }
    } else {
        constexpr int DC = CIN / 32;           // CIN is a multiple of 32 here
        // chunks 0..DC-1: neighbor-minus-center channels
#pragma unroll
        for (int kc = 0; kc < DC; ++kc) {
            BF16x16 Bt;
            const float4* pj = (const float4*)(xj + kc * 32 + bkbase);
            const float4* pn = (const float4*)(xn + kc * 32 + bkbase);
#pragma unroll
            for (int q = 0; q < 4; ++q) {
                float4 a = pj[q], c = pn[q];
                Bt.s[4 * q + 0] = f32_to_bf16(a.x - c.x);
                Bt.s[4 * q + 1] = f32_to_bf16(a.y - c.y);
                Bt.s[4 * q + 2] = f32_to_bf16(a.z - c.z);
                Bt.s[4 * q + 3] = f32_to_bf16(a.w - c.w);
            }
#pragma unroll
            for (int t = 0; t < MT; ++t) {
                const uint4* ap = apLane + (size_t)((mtile0 + t) * KCH + kc) * 64;
                BF16x16 A;  A.q[0] = ap[0];  A.q[1] = ap[1];
                acc[t] = __builtin_amdgcn_wmma_f32_16x16x32_bf16(
                             false, A.v, false, Bt.v, (short)0, acc[t], false, false);
            }
        }
        // chunks DC..2*DC-1: center channels
#pragma unroll
        for (int kc = 0; kc < DC; ++kc) {
            BF16x16 Bt;
            const float4* pc = (const float4*)(xn + kc * 32 + bkbase);
#pragma unroll
            for (int q = 0; q < 4; ++q) {
                float4 a = pc[q];
                Bt.s[4 * q + 0] = f32_to_bf16(a.x);
                Bt.s[4 * q + 1] = f32_to_bf16(a.y);
                Bt.s[4 * q + 2] = f32_to_bf16(a.z);
                Bt.s[4 * q + 3] = f32_to_bf16(a.w);
            }
#pragma unroll
            for (int t = 0; t < MT; ++t) {
                const uint4* ap = apLane + (size_t)((mtile0 + t) * KCH + DC + kc) * 64;
                BF16x16 A;  A.q[0] = ap[0];  A.q[1] = ap[1];
                acc[t] = __builtin_amdgcn_wmma_f32_16x16x32_bf16(
                             false, A.v, false, Bt.v, (short)0, acc[t], false, false);
            }
        }
    }

    // C/D layout: vgpr r, lane L -> channel mtile*16 + r + (L/16)*8, pixel L%16
    if (mode == 0) {
#pragma unroll
        for (int t = 0; t < MT; ++t) {
#pragma unroll
            for (int r = 0; r < 8; ++r) {
                float s1 = acc[t][r];
                float s2 = s1 * s1;
#pragma unroll
                for (int off = 1; off < 16; off <<= 1) {
                    s1 += __shfl_xor(s1, off, 32);
                    s2 += __shfl_xor(s2, off, 32);
                }
                if ((lane & 15) == 0) {
                    int oc = (mtile0 + t) * 16 + r + (lane >> 4) * 8;
                    atomicAdd(&sums[oc], s1);
                    atomicAdd(&sums[COUT + oc], s2);
                }
            }
        }
    } else {
#pragma unroll
        for (int t = 0; t < MT; ++t) {
#pragma unroll
            for (int r = 0; r < 8; ++r) {
                int oc = (mtile0 + t) * 16 + r + (lane >> 4) * 8;
                float yv = acc[t][r] * scA[oc] + scC[oc];
                yv = (yv >= 0.f) ? yv : 0.2f * yv;
                unsigned e = enc_f32(yv);
                if (mode == 2) atomicMax(&outEnc[b * COUT + oc], e);
                else           atomicMax(&hEnc[((size_t)b * NPTS + n) * COUT + oc], e);
            }
        }
    }
}

// fold BN stats into per-channel affine: y' = a*y + c
__global__ void bn_stats(const float* __restrict__ sums, const float* __restrict__ g,
                         const float* __restrict__ beta, float* __restrict__ scA,
                         float* __restrict__ scC, int Cout, float invCnt) {
    int o = blockIdx.x * blockDim.x + threadIdx.x;
    if (o < Cout) {
        float mean = sums[o] * invCnt;
        float var  = sums[Cout + o] * invCnt - mean * mean;
        float a    = g[o] * rsqrtf(var + 1e-5f);
        scA[o] = a;
        scC[o] = beta[o] - mean * a;
    }
}

__global__ void fill_u32(unsigned* __restrict__ p, unsigned v, int nwords) {
    int i = blockIdx.x * blockDim.x + threadIdx.x;
    if (i < nwords) p[i] = v;
}

__global__ void decode_kernel(const unsigned* __restrict__ e, float* __restrict__ f, int nelem) {
    int i = blockIdx.x * blockDim.x + threadIdx.x;
    if (i < nelem) f[i] = dec_f32(e[i]);
}

// ---------------------------------------------------------------------------
struct Ws {
    float*    h0;  float* h1;  float* xT0;
    unsigned* hEnc; int* idxb; float* sqn;
    float*    sums; float* scA; float* scC;
    unsigned* apack; unsigned* outEnc;
};

template <int CIN, int COUT, int MT>
static void run_layer(const float* xin, const float* W, const float* g, const float* beta,
                      bool finalLayer, float* hout, float* outF, const Ws& w,
                      hipStream_t stream) {
    constexpr int TWOC = 2 * CIN;
    constexpr int KCH  = (TWOC + 31) / 32;

    sqnorm_kernel<CIN><<<(NB * NPTS) / 256, 256, 0, stream>>>(xin, w.sqn);
    knn_kernel<CIN><<<(NB * NPTS) / 64, 64, 0, stream>>>(xin, w.sqn, w.idxb);

    int packThreads = (COUT / 16) * KCH * 32;
    pack_w<CIN, COUT><<<(packThreads + 255) / 256, 256, 0, stream>>>(W, w.apack);

    fill_u32<<<1, 512, 0, stream>>>((unsigned*)w.sums, 0u, 2 * COUT);

    dim3 grid(NPIX / 16, COUT / (16 * MT), NB);
    edgeconv_kernel<CIN, COUT, MT><<<grid, 32, 0, stream>>>(
        xin, w.idxb, w.apack, w.scA, w.scC, w.sums, w.hEnc, w.outEnc, 0);

    float invCnt = 1.0f / ((float)NB * NPTS * KNN);
    bn_stats<<<1, 256, 0, stream>>>(w.sums, g, beta, w.scA, w.scC, COUT, invCnt);

    if (!finalLayer) {
        int nwords = NB * NPTS * COUT;
        fill_u32<<<(nwords + 255) / 256, 256, 0, stream>>>(w.hEnc, 0u, nwords);
        edgeconv_kernel<CIN, COUT, MT><<<grid, 32, 0, stream>>>(
            xin, w.idxb, w.apack, w.scA, w.scC, w.sums, w.hEnc, w.outEnc, 1);
        decode_kernel<<<(nwords + 255) / 256, 256, 0, stream>>>(w.hEnc, hout, nwords);
    } else {
        fill_u32<<<(4096 + 255) / 256, 256, 0, stream>>>(w.outEnc, 0u, 4096);
        edgeconv_kernel<CIN, COUT, MT><<<grid, 32, 0, stream>>>(
            xin, w.idxb, w.apack, w.scA, w.scC, w.sums, w.hEnc, w.outEnc, 2);
        decode_kernel<<<(4096 + 255) / 256, 256, 0, stream>>>(w.outEnc, outF, 4096);
    }
}

extern "C" void kernel_launch(void* const* d_in, const int* in_sizes, int n_in,
                              void* d_out, int out_size, void* d_ws, size_t ws_size,
                              hipStream_t stream) {
    (void)in_sizes; (void)n_in; (void)out_size; (void)ws_size;

    const float* x0 = (const float*)d_in[0];
    const float* Wp[4] = {(const float*)d_in[1], (const float*)d_in[4],
                          (const float*)d_in[7], (const float*)d_in[10]};
    const float* gp[4] = {(const float*)d_in[2], (const float*)d_in[5],
                          (const float*)d_in[8], (const float*)d_in[11]};
    const float* bp[4] = {(const float*)d_in[3], (const float*)d_in[6],
                          (const float*)d_in[9], (const float*)d_in[12]};

    char* ws = (char*)d_ws;
    size_t off = 0;
    auto alloc = [&](size_t bytes) -> void* {
        void* p = ws + off;
        off += (bytes + 255) & ~(size_t)255;
        return p;
    };
    Ws w;
    w.h0     = (float*)   alloc((size_t)NB * 256 * NPTS * 4);
    w.h1     = (float*)   alloc((size_t)NB * 256 * NPTS * 4);
    w.hEnc   = (unsigned*)alloc((size_t)NB * 256 * NPTS * 4);
    w.xT0    = (float*)   alloc((size_t)NB * NPTS * 4 * 4);
    w.idxb   = (int*)     alloc((size_t)NB * NPTS * KNN * 4);
    w.sqn    = (float*)   alloc((size_t)NB * NPTS * 4);
    w.sums   = (float*)   alloc(512 * 4);
    w.scA    = (float*)   alloc(256 * 4);
    w.scC    = (float*)   alloc(256 * 4);
    w.apack  = (unsigned*)alloc((size_t)16 * 16 * 32 * 8 * 4);   // max frags
    w.outEnc = (unsigned*)alloc(4096 * 4);

    // layer 0 input -> point-major
    transpose_x0<<<(NB * NPTS * 4) / 256, 256, 0, stream>>>(x0, w.xT0);

    float* outF = (float*)d_out;
    run_layer<4,   64,  4>(w.xT0, Wp[0], gp[0], bp[0], false, w.h0, outF, w, stream);
    run_layer<64,  64,  4>(w.h0,  Wp[1], gp[1], bp[1], false, w.h1, outF, w, stream);
    run_layer<64,  128, 8>(w.h1,  Wp[2], gp[2], bp[2], false, w.h0, outF, w, stream);
    run_layer<128, 256, 8>(w.h0,  Wp[3], gp[3], bp[3], true,  nullptr, outF, w, stream);
}
